// MCW_RNN_43559558316166
// MI455X (gfx1250) — compile-verified, matrix operand order checked by reading, not run on hardware
//
#include <hip/hip_runtime.h>
#include <cmath>

// ---------------------------------------------------------------------------
// MCW clockwork RNN for MI455X (gfx1250, wave32, WMMA).
//
// Shape (fixed by reference setup): B=2048, T=512, DX=32, DY=4, DH1=256.
//   * one block per (16-row batch tile, head): grid = (B/16)*DY = 512
//   * recurrent weights [Wx;Wh] (288x256 f16) pinned in VGPRs as WMMA
//     B-fragments (8 waves x 2 N-tiles x 9 K-blocks = 144 VGPRs/wave)
//   * state [x_eff | h] double-buffered in LDS, buffer stride 8192 halves so
//     buffer toggling is one XOR on precomputed flat offsets; padded rows
//     (296 halves) give conflict-free ds_load_b128 A-fragments
//   * each wave carries its h tiles in registers -> every tile rewritten each
//     step -> ONE barrier per step (no WAR on the read buffer)
//   * A-fragment loads pipelined 2 k-blocks ahead; schedule pinned with
//     sched_group_barrier so waits become s_wait_dscnt 0x2 (not 0x0)
//   * clockwork flags via carried wrap counters (no runtime-div SALU chains)
//   * x(t+1) global loads issued before the WMMA chain, consumed after it
// ---------------------------------------------------------------------------

typedef __attribute__((ext_vector_type(16))) _Float16 v16h;
typedef __attribute__((ext_vector_type(8)))  _Float16 v8h;
typedef __attribute__((ext_vector_type(8)))  float    v8f;

#define T_STEPS   512
#define DXC       32
#define DH1C      256
#define DYC       4
#define KBLK      9            // (DX + DH1) / 32
#define ROWH      296          // 288 halves + 8 pad -> 592B row stride (16-aligned)
#define BUFH      8192         // halves per buffer (pow2 -> XOR toggling)
#define BLOCK_T   256

#if __has_builtin(__builtin_amdgcn_tanhf)
#define FAST_TANH(x) __builtin_amdgcn_tanhf(x)
#elif __has_builtin(__builtin_amdgcn_tanh_f32)
#define FAST_TANH(x) __builtin_amdgcn_tanh_f32(x)
#else
#define FAST_TANH(x) tanhf(x)
#endif

#if __has_builtin(__builtin_amdgcn_sched_group_barrier)
#define SGB(mask, cnt) __builtin_amdgcn_sched_group_barrier((mask), (cnt), 0)
#else
#define SGB(mask, cnt)
#endif
// sched_group_barrier masks: 0x8 = MFMA/WMMA, 0x100 = DS read

// Load one 16x32 f16 A-fragment for this lane from the LDS state buffer.
// A layout (ISA 7.12.2): lane<16 : M=lane,    halves 0..7 = K 0..7,  8..15 = K 16..23
//                        lane>=16: M=lane-16, halves 0..7 = K 8..15, 8..15 = K 24..31
__device__ __forceinline__ v16h lds_afrag(const _Float16* rowBase, int byteOff) {
  v16h r;
  const float4* p0 = (const float4*)((const char*)rowBase + byteOff);
  const float4* p1 = (const float4*)((const char*)rowBase + byteOff + 32);
  ((float4*)&r)[0] = *p0;     // ds_load_b128
  ((float4*)&r)[1] = *p1;     // ds_load_b128
  return r;
}

__global__ __launch_bounds__(BLOCK_T)
void mcw_rnn_kernel(const float* __restrict__ x,
                    const float* __restrict__ w_x,
                    const float* __restrict__ w_h,
                    const float* __restrict__ bvec,
                    const float* __restrict__ mlp_w1,
                    const float* __restrict__ mlp_b1,
                    const float* __restrict__ mlp_w2,
                    const float* __restrict__ mlp_b2,
                    const int* __restrict__ p_len1,
                    const int* __restrict__ p_len2,
                    const int* __restrict__ p_r2,
                    const int* __restrict__ p_r3,
                    float* __restrict__ out)
{
  __shared__ __align__(16) _Float16 HXf[2 * BUFH];  // 2 x [16 rows x ROWH halves]
  __shared__ float red[BLOCK_T];

  const int tid  = threadIdx.x;
  const int wv   = tid >> 5;        // wave id 0..7
  const int lane = tid & 31;
  const int li   = lane & 15;
  const int hi   = lane >> 4;       // half-wave select

  const int y  = blockIdx.x & (DYC - 1);
  const int bt = blockIdx.x >> 2;
  const int b0 = bt * 16;

  const int len1  = p_len1[0];
  const int len12 = len1 + p_len2[0];
  const int r2    = p_r2[0];
  const int r3    = p_r3[0];
  const int mult  = DH1C / DXC;
  const int Hb1   = len1 * mult;                 // 128: always-update boundary
  const int Hb2   = Hb1 + p_len2[0] * mult;      // 192: x2_rate boundary

  // ---- Preload B-fragments (weights) into VGPRs: 2 N-tiles x 9 K-blocks ----
  // Assumed B layout mirrors A with N in place of M.
  v16h Bf[2][KBLK];
  float bias[2];
  const int nt0 = 2 * wv;
#pragma unroll
  for (int j = 0; j < 2; ++j) {
    const int nt = nt0 + j;
    const int n  = nt * 16 + li;
    bias[j] = bvec[y * DH1C + n];
#pragma unroll
    for (int kb = 0; kb < KBLK; ++kb) {
      v16h bf;
#pragma unroll
      for (int h = 0; h < 16; ++h) {
        const int kl = hi * 8 + (h < 8 ? h : h + 8);   // local K in [0,32)
        const int K  = kb * 32 + kl;                   // concat index over [x|h]
        const float v = (K < DXC)
            ? w_x[(size_t)y * DXC * DH1C + (size_t)K * DH1C + n]
            : w_h[(size_t)y * DH1C * DH1C + (size_t)(K - DXC) * DH1C + n];
        bf[h] = (_Float16)v;
      }
      Bf[j][kb] = bf;
    }
  }

  // Per-tile clockwork region: 0 = always, 1 = x2_rate, 2 = x3_rate
  int region[2];
#pragma unroll
  for (int j = 0; j < 2; ++j) {
    const int nb = (nt0 + j) * 16;
    region[j] = (nb < Hb1) ? 0 : (nb < Hb2 ? 1 : 2);
  }

  // Register-carried h tiles (8 halves per lane per tile), start at h0 = 0.
  v8h hreg[2];
#pragma unroll
  for (int j = 0; j < 2; ++j)
#pragma unroll
    for (int r = 0; r < 8; ++r) hreg[j][r] = (_Float16)0.0f;

  // ---- Precomputed flat LDS offsets (halves); XOR BUFH toggles buffers ----
  int rdRow = li * ROWH;                       // read rows in buffer 0 first
  int wrX0, wrX1;                              // x-stage slots (next buffer)
  int xr0, xk0, xr1, xk1;
  {
    xr0 = tid >> 5;             xk0 = tid & 31;
    xr1 = (tid + BLOCK_T) >> 5; xk1 = (tid + BLOCK_T) & 31;
    wrX0 = BUFH + xr0 * ROWH + xk0;
    wrX1 = BUFH + xr1 * ROWH + xk1;
  }
  int wrH[2];
#pragma unroll
  for (int j = 0; j < 2; ++j)
    wrH[j] = BUFH + (8 * hi) * ROWH + DXC + (nt0 + j) * 16 + li;

  // Incrementally advanced global x pointers (start at t = 1).
  const float* xp0 = x + (size_t)(b0 + xr0) * (T_STEPS * DXC) + DXC + xk0;
  const float* xp1 = x + (size_t)(b0 + xr1) * (T_STEPS * DXC) + DXC + xk1;

  // ---- Init buffer 0: zero h region, stage masked x(0) ----
  for (int i = tid; i < 16 * ROWH; i += BLOCK_T) HXf[i] = (_Float16)0.0f;
  {
    const bool u2 = (1 % r2) == 0, u3 = (1 % r3) == 0;
    const float v0 = xp0[-DXC], v1 = xp1[-DXC];
    const bool m0 = (xk0 < len1) | (u2 & (xk0 < len12)) | u3;
    const bool m1 = (xk1 < len1) | (u2 & (xk1 < len12)) | u3;
    HXf[wrX0 - BUFH] = (_Float16)(m0 ? v0 : 0.0f);
    HXf[wrX1 - BUFH] = (_Float16)(m1 ? v1 : 0.0f);
  }

  // Clockwork wrap counters: m2 = (t+1) % r2, m3 = (t+1) % r3 (carried).
  int m2 = 1 % r2, m3 = 1 % r3;

  // ---------------------------- recurrence ---------------------------------
  for (int t = 0; t < T_STEPS; ++t) {
    __syncthreads();            // current buffer (x(t) + h(t)) fully written

    const bool upd2 = (m2 == 0), upd3 = (m3 == 0);
    int m2n = m2 + 1; if (m2n == r2) m2n = 0;   // (t+2) % r2
    int m3n = m3 + 1; if (m3n == r3) m3n = 0;   // (t+2) % r3
    const bool u2n = (m2n == 0), u3n = (m3n == 0);

    // Issue x(t+1) global loads early; consume after the WMMA chain.
    const bool have_nx = (t + 1) < T_STEPS;
    float xv0 = 0.0f, xv1 = 0.0f;
    if (have_nx) {
      xv0 = *xp0;                                   // global_load_b32
      xv1 = *xp1;
      if (t + 2 < T_STEPS) {
        __builtin_prefetch(xp0 + DXC, 0, 1);        // global_prefetch_b8
        __builtin_prefetch(xp1 + DXC, 0, 1);
      }
    }

    // pre = [x|h] @ [Wx;Wh] + b; A-fragments pipelined 2 k-blocks ahead.
    v8f acc0, acc1;
#pragma unroll
    for (int r = 0; r < 8; ++r) { acc0[r] = bias[0]; acc1[r] = bias[1]; }

    const _Float16* rowBase = &HXf[rdRow];
    v16h fa = lds_afrag(rowBase, hi * 16);                 // kb 0
    v16h fb = lds_afrag(rowBase, 64 + hi * 16);            // kb 1
#pragma unroll
    for (int kb = 0; kb < KBLK; ++kb) {
      const v16h a = (kb & 1) ? fb : fa;
      acc0 = __builtin_amdgcn_wmma_f32_16x16x32_f16(
                 false, a, false, Bf[0][kb], (short)0, acc0, false, false);
      acc1 = __builtin_amdgcn_wmma_f32_16x16x32_f16(
                 false, a, false, Bf[1][kb], (short)0, acc1, false, false);
      if (kb + 2 < KBLK) {
        const v16h nf = lds_afrag(rowBase, (kb + 2) * 64 + hi * 16);
        if (kb & 1) fb = nf; else fa = nf;
      }
    }

    // Pin the software pipeline:
    //   DS_READ x4 (kb0,kb1) ; [ WMMA x2 (kb) ; DS_READ x2 (kb+2) ] x7 ;
    //   WMMA x2 (kb7) ; WMMA x2 (kb8)
    // Forces the kb+1 loads in flight before kb's WMMAs issue -> two live
    // fragment buffers -> s_wait_dscnt 0x2 instead of full drains.
    SGB(0x100, 4);
#pragma unroll
    for (int g = 0; g < KBLK - 2; ++g) { SGB(0x8, 2); SGB(0x100, 2); }
    SGB(0x8, 2);
    SGB(0x8, 2);

    // Stage masked x(t+1) into the next buffer (loads already in flight).
    if (have_nx) {
      const bool mm0 = (xk0 < len1) | (u2n & (xk0 < len12)) | u3n;
      const bool mm1 = (xk1 < len1) | (u2n & (xk1 < len12)) | u3n;
      HXf[wrX0] = (_Float16)(mm0 ? xv0 : 0.0f);
      HXf[wrX1] = (_Float16)(mm1 ? xv1 : 0.0f);
      xp0 += DXC; xp1 += DXC;
    }

    // Clockwork update of register-carried h, then unconditional store to
    // the next buffer (D layout: element r -> (M = r + 8*hi, N = nt*16+li)).
#pragma unroll
    for (int j = 0; j < 2; ++j) {
      const bool upd = (region[j] == 0) | ((region[j] == 1) & upd2)
                                        | ((region[j] == 2) & upd3);
      const v8f acc = j == 0 ? acc0 : acc1;
      if (upd) {
#pragma unroll
        for (int r = 0; r < 8; ++r) hreg[j][r] = (_Float16)FAST_TANH(acc[r]);
      }
#pragma unroll
      for (int r = 0; r < 8; ++r)
        HXf[wrH[j] + r * ROWH] = hreg[j][r];     // ds_store_b16 (imm offsets)
    }

    // Toggle buffers (single XOR each) and advance counters.
    rdRow ^= BUFH; wrX0 ^= BUFH; wrX1 ^= BUFH;
    wrH[0] ^= BUFH; wrH[1] ^= BUFH;
    m2 = m2n; m3 = m3n;
  }
  __syncthreads();

  // ------------------------------ MLP head ---------------------------------
  // Final h lives in buffer (T_STEPS & 1): T even -> buffer 0.
  const _Float16* HF = &HXf[(T_STEPS & 1) * BUFH];
  {
    const int r  = tid & 15;
    const int cb = tid >> 4;
    float partial = 0.0f;
    for (int cc = 0; cc < 16; ++cc) {
      const int c = cb * 16 + cc;
      float s = mlp_b1[y * DH1C + c];
      for (int k = 0; k < DH1C; ++k)
        s += (float)HF[r * ROWH + DXC + k] *
             mlp_w1[(size_t)y * DH1C * DH1C + (size_t)k * DH1C + c];
      s = fmaxf(s, 0.0f);
      partial += s * mlp_w2[y * DH1C + c];
    }
    red[tid] = partial;
  }
  __syncthreads();
  if (tid < 16) {
    float s = mlp_b2[y];
#pragma unroll
    for (int j = 0; j < 16; ++j) s += red[tid + (j << 4)];
    out[(size_t)(b0 + tid) * DYC + y] = s;
  }
}

extern "C" void kernel_launch(void* const* d_in, const int* in_sizes, int n_in,
                              void* d_out, int out_size, void* d_ws, size_t ws_size,
                              hipStream_t stream) {
  const float* x      = (const float*)d_in[0];
  const float* w_x    = (const float*)d_in[1];
  const float* w_h    = (const float*)d_in[2];
  const float* bvec   = (const float*)d_in[3];
  const float* mlp_w1 = (const float*)d_in[4];
  const float* mlp_b1 = (const float*)d_in[5];
  const float* mlp_w2 = (const float*)d_in[6];
  const float* mlp_b2 = (const float*)d_in[7];
  const int*   len1   = (const int*)d_in[8];
  const int*   len2   = (const int*)d_in[9];
  // d_in[10] = len_x3 (implied by DX - len1 - len2; unused)
  const int*   r2     = (const int*)d_in[11];
  const int*   r3     = (const int*)d_in[12];
  float*       out    = (float*)d_out;

  const int B = in_sizes[0] / (T_STEPS * DXC);   // 2048
  const int grid = (B / 16) * DYC;               // 512 blocks

  mcw_rnn_kernel<<<grid, BLOCK_T, 0, stream>>>(
      x, w_x, w_h, bvec, mlp_w1, mlp_b1, mlp_w2, mlp_b2,
      len1, len2, r2, r3, out);
}
